// PerformerAttentionMinimal_48696339202316
// MI455X (gfx1250) — matrix operand order repaired and checked by reading, not compiled, not run on hardware
//
#include <hip/hip_runtime.h>

typedef __bf16 bf16_t;
typedef bf16_t v16bf __attribute__((ext_vector_type(16)));
typedef bf16_t v8bf  __attribute__((ext_vector_type(8)));
typedef float  v8f   __attribute__((ext_vector_type(8)));

#define WMMA_BF16(a, b, c) \
  __builtin_amdgcn_wmma_f32_16x16x32_bf16(false, (a), false, (b), (short)0, (c), false, false)

__device__ __forceinline__ void vzero(v8f& v) {
#pragma unroll
  for (int r = 0; r < 8; ++r) v[r] = 0.f;
}

// ---- async global<->LDS helpers (CDNA5, ASYNCcnt) --------------------------
typedef __attribute__((address_space(3))) void as3_void;

__device__ __forceinline__ unsigned lds_off_u32(const void* p) {
  // generic -> AS3 (addrspacecast) -> 32-bit LDS offset
  return (unsigned)(unsigned long long)(as3_void*)p;
}
__device__ __forceinline__ void async_g2l_b128(unsigned lds_off, const void* g) {
  asm volatile("global_load_async_to_lds_b128 %0, %1, off"
               :: "v"(lds_off), "v"(g) : "memory");
}
__device__ __forceinline__ void async_l2g_b128(const void* g, unsigned lds_off) {
  asm volatile("global_store_async_from_lds_b128 %0, %1, off"
               :: "v"(g), "v"(lds_off) : "memory");
}
__device__ __forceinline__ void wait_async0() {
  asm volatile("s_wait_asynccnt 0x0" ::: "memory");
}

// A fragment: 16(M) x 32(K) bf16 tile, row-major in LDS with row stride `stride`.
// ISA: lane L holds row (L&15), K in {(L/16)*8 .. +7} and {16+(L/16)*8 .. +7}.
__device__ __forceinline__ v16bf frag_a(const bf16_t* base, int stride, int lane) {
  const int row = lane & 15;
  const int kb  = (lane >> 4) << 3;
  const bf16_t* p = base + row * stride + kb;
  v8bf lo = *(const v8bf*)p;
  v8bf hi = *(const v8bf*)(p + 16);
  return __builtin_shufflevector(lo, hi, 0, 1, 2, 3, 4, 5, 6, 7, 8, 9, 10, 11, 12, 13, 14, 15);
}

// B fragment from a TRANSPOSED B tile (BT row-major [N][K], row stride `stride`).
// ISA: lane L holds col (L&15), K in {(L/16)*8 .. +7} and {16+(L/16)*8 .. +7}.
__device__ __forceinline__ v16bf frag_b(const bf16_t* base, int stride, int lane) {
  const int col = lane & 15;
  const int kb  = (lane >> 4) << 3;
  const bf16_t* p = base + col * stride + kb;
  v8bf lo = *(const v8bf*)p;
  v8bf hi = *(const v8bf*)(p + 16);
  return __builtin_shufflevector(lo, hi, 0, 1, 2, 3, 4, 5, 6, 7, 8, 9, 10, 11, 12, 13, 14, 15);
}

__device__ __forceinline__ void cvt4(bf16_t* p, float4 f) {
  p[0] = (bf16_t)f.x; p[1] = (bf16_t)f.y; p[2] = (bf16_t)f.z; p[3] = (bf16_t)f.w;
}

// ---------------------------------------------------------------------------
// K1: qkv GEMM. q,k thirds -> qkv[32768 x 1536] bf16 (LDS transpose tile, then
//     async LDS->global b128 stores). v third -> vT[bh][64 d][16384 n] bf16.
// ---------------------------------------------------------------------------
__global__ __launch_bounds__(256) void k_qkv(const float* __restrict__ X,
                                             const float* __restrict__ W,
                                             bf16_t* __restrict__ Y,
                                             bf16_t* __restrict__ vT) {
  const int tid = threadIdx.x, lane = tid & 31, wave = tid >> 5;
  const int wm = wave & 3, wn = wave >> 2;                 // 4x2 waves, wave tile 32x64
  const long m0 = (long)blockIdx.y * 128;
  const int  n0 = blockIdx.x * 128;
  __shared__ __attribute__((aligned(128))) bf16_t sA[128 * 32];
  __shared__ __attribute__((aligned(128))) bf16_t sB[128 * 32];
  __shared__ __attribute__((aligned(128))) bf16_t sOut[128 * 128];
  v8f acc[2][4];
#pragma unroll
  for (int i = 0; i < 2; ++i)
#pragma unroll
    for (int j = 0; j < 4; ++j) vzero(acc[i][j]);

  for (int k0 = 0; k0 < 512; k0 += 32) {
    __syncthreads();
    for (int idx = tid; idx < 128 * 8; idx += 256) {
      int r = idx >> 3, c4 = (idx & 7) * 4;
      cvt4(sA + r * 32 + c4, *(const float4*)(X + (m0 + r) * 512 + k0 + c4));
      cvt4(sB + r * 32 + c4, *(const float4*)(W + (long)(n0 + r) * 512 + k0 + c4));
    }
    if (k0 + 32 < 512) {                                   // prefetch next K tile
      if (tid < 128) __builtin_prefetch(X + (m0 + tid) * 512 + k0 + 32, 0, 1);
      else           __builtin_prefetch(W + (long)(n0 + tid - 128) * 512 + k0 + 32, 0, 1);
    }
    __syncthreads();
    v16bf a[2], b[4];
#pragma unroll
    for (int i = 0; i < 2; ++i) a[i] = frag_a(sA + (wm * 32 + i * 16) * 32, 32, lane);
#pragma unroll
    for (int j = 0; j < 4; ++j) b[j] = frag_b(sB + (wn * 64 + j * 16) * 32, 32, lane);
#pragma unroll
    for (int i = 0; i < 2; ++i)
#pragma unroll
      for (int j = 0; j < 4; ++j) acc[i][j] = WMMA_BF16(a[i], b[j], acc[i][j]);
  }

  if (n0 < 1024) {
    // q/k: scatter to LDS (transpose within tile), then async LDS->global b128
#pragma unroll
    for (int i = 0; i < 2; ++i)
#pragma unroll
      for (int j = 0; j < 4; ++j) {
        int colL = wn * 64 + j * 16 + (lane & 15);
        int nb   = wm * 32 + i * 16 + ((lane >> 4) << 3);
#pragma unroll
        for (int r = 0; r < 8; ++r) sOut[(nb + r) * 128 + colL] = (bf16_t)acc[i][j][r];
      }
    __syncthreads();
    const unsigned so = lds_off_u32(sOut);
    for (int idx = tid; idx < 128 * 16; idx += 256) {
      int r = idx >> 4, c8 = (idx & 15) * 8;
      async_l2g_b128(Y + (m0 + r) * 1536 + n0 + c8, so + (unsigned)(r * 128 + c8) * 2);
    }
    // s_endpgm performs an implicit wait-idle; stores drain before kernel end.
  } else {
    // v: packed transposed stores -> vT[bh][d][n]
    const int b = (int)(m0 >> 14);
    const int nbase0 = (int)(m0 & 16383);
#pragma unroll
    for (int i = 0; i < 2; ++i)
#pragma unroll
      for (int j = 0; j < 4; ++j) {
        int col = n0 + wn * 64 + j * 16 + (lane & 15);
        int h = (col - 1024) >> 6, d = col & 63;
        int nb = nbase0 + wm * 32 + i * 16 + ((lane >> 4) << 3);
        v8bf t;
#pragma unroll
        for (int r = 0; r < 8; ++r) t[r] = (bf16_t)acc[i][j][r];
        *(v8bf*)(vT + ((long)((b * 8 + h) * 64 + d)) * 16384 + nb) = t;
      }
  }
}

// ---------------------------------------------------------------------------
// K2: per (b,h) chunk: E = exp(K @ proj^T); k_sum and partial context = E^T @ V
//     entirely on-chip. K/V^T tiles staged with async DMA. Grid: (16 bh, 16).
// ---------------------------------------------------------------------------
__global__ __launch_bounds__(256) void k_kfeat(const bf16_t* __restrict__ qkv,
                                               const bf16_t* __restrict__ vT,
                                               const float* __restrict__ proj,
                                               float* __restrict__ ctx_part,
                                               float* __restrict__ ksum_part) {
  const int bh = blockIdx.x, chunk = blockIdx.y;
  const int tid = threadIdx.x, lane = tid & 31, wave = tid >> 5;
  const int wm1 = wave & 3, wn1 = wave >> 2;               // stage1: 4x2, wave tile 32n x 128f
  const int h = bh & 7;
  const long rowbase = (long)(bh >> 3) * 16384;
  const bf16_t* Kg  = qkv + rowbase * 1536 + 512 + h * 64;
  const bf16_t* VTg = vT + (long)bh * 64 * 16384;

  __shared__ __attribute__((aligned(128))) bf16_t sProj[256 * 64];  // proj rows = B^T
  __shared__ __attribute__((aligned(128))) bf16_t sA[128 * 64];     // K tile [n][d]
  __shared__ __attribute__((aligned(128))) bf16_t sET[256 * 128];   // exp(k') [f][n]
  __shared__ __attribute__((aligned(128))) bf16_t sVT[64 * 128];    // V^T tile [d][n]
  __shared__ float sKsum[256];

  const unsigned sA_off  = lds_off_u32(sA);
  const unsigned sVT_off = lds_off_u32(sVT);

  for (int idx = tid; idx < 256 * 16; idx += 256) {
    int r = idx >> 4, c4 = (idx & 15) * 4;
    cvt4(sProj + r * 64 + c4, *(const float4*)(proj + r * 64 + c4));
  }
  sKsum[tid] = 0.f;

  v8f c2[2][4];                                            // stage2 acc: 32f x 64d per wave
#pragma unroll
  for (int i = 0; i < 2; ++i)
#pragma unroll
    for (int j = 0; j < 4; ++j) vzero(c2[i][j]);

  for (int t = 0; t < 8; ++t) {
    const long n0 = (long)(chunk * 8 + t) * 128;
    __syncthreads();                                       // buffers free (all waves past reads)
    for (int idx = tid; idx < 128 * 8; idx += 256) {       // K tile rows, async DMA
      int r = idx >> 3, c8 = (idx & 7) * 8;
      async_g2l_b128(sA_off + (unsigned)(r * 64 + c8) * 2, Kg + (n0 + r) * 1536 + c8);
    }
    for (int idx = tid; idx < 64 * 16; idx += 256) {       // V^T tile rows, async DMA
      int r = idx >> 4, c8 = (idx & 15) * 8;
      async_g2l_b128(sVT_off + (unsigned)(r * 128 + c8) * 2, VTg + (long)r * 16384 + n0 + c8);
    }
    wait_async0();
    __syncthreads();                                       // all waves' DMA visible

    // stage 1: E = K @ proj^T   (M=128 n, N=256 f, K=64)
    v8f e[2][8];
#pragma unroll
    for (int i = 0; i < 2; ++i)
#pragma unroll
      for (int j = 0; j < 8; ++j) vzero(e[i][j]);
#pragma unroll
    for (int k0 = 0; k0 < 64; k0 += 32) {
      v16bf a[2];
#pragma unroll
      for (int i = 0; i < 2; ++i) a[i] = frag_a(sA + (wm1 * 32 + i * 16) * 64 + k0, 64, lane);
#pragma unroll
      for (int j = 0; j < 8; ++j) {
        v16bf b = frag_b(sProj + (wn1 * 128 + j * 16) * 64 + k0, 64, lane);
#pragma unroll
        for (int i = 0; i < 2; ++i) e[i][j] = WMMA_BF16(a[i], b, e[i][j]);
      }
    }
    // exp() and packed transposed write (8 consecutive n per lane -> b128)
#pragma unroll
    for (int i = 0; i < 2; ++i)
#pragma unroll
      for (int j = 0; j < 8; ++j) {
        int nb = wm1 * 32 + i * 16 + ((lane >> 4) << 3);
        int f  = wn1 * 128 + j * 16 + (lane & 15);
        v8bf tpk;
#pragma unroll
        for (int r = 0; r < 8; ++r) tpk[r] = (bf16_t)__expf(e[i][j][r]);
        *(v8bf*)(sET + f * 128 + nb) = tpk;
      }
    __syncthreads();

    // k_sum: thread tid owns feature f = tid
    {
      float s = 0.f;
      const bf16_t* p = sET + tid * 128;
      for (int q = 0; q < 128; q += 8) {
        v8bf v = *(const v8bf*)(p + q);
#pragma unroll
        for (int u = 0; u < 8; ++u) s += (float)v[u];
      }
      sKsum[tid] += s;
    }

    // stage 2: context += E^T @ V  (M=256 f, N=64 d, K=128 n); wave owns 32 f rows
#pragma unroll
    for (int k0 = 0; k0 < 128; k0 += 32) {
      v16bf a[2], b[4];
#pragma unroll
      for (int i = 0; i < 2; ++i) a[i] = frag_a(sET + (wave * 32 + i * 16) * 128 + k0, 128, lane);
#pragma unroll
      for (int j = 0; j < 4; ++j) b[j] = frag_b(sVT + (j * 16) * 128 + k0, 128, lane);
#pragma unroll
      for (int i = 0; i < 2; ++i)
#pragma unroll
        for (int j = 0; j < 4; ++j) c2[i][j] = WMMA_BF16(a[i], b[j], c2[i][j]);
    }
  }

  float* cp = ctx_part + ((long)chunk * 16 + bh) * (256 * 64);
#pragma unroll
  for (int i = 0; i < 2; ++i)
#pragma unroll
    for (int j = 0; j < 4; ++j)
#pragma unroll
      for (int r = 0; r < 8; ++r) {
        int f = wave * 32 + i * 16 + r + ((lane >> 4) << 3);
        int d = j * 16 + (lane & 15);
        cp[f * 64 + d] = c2[i][j][r];
      }
  ksum_part[((long)chunk * 16 + bh) * 256 + tid] = sKsum[tid];
}

// ---------------------------------------------------------------------------
// K3: reduce 16 chunk partials -> contextT (bh, 64 d, 256 f) and k_sum (bh, 256)
// ---------------------------------------------------------------------------
__global__ __launch_bounds__(256) void k_reduce(const float* __restrict__ ctx_part,
                                                const float* __restrict__ ksum_part,
                                                float* __restrict__ ctxT,
                                                float* __restrict__ ksum) {
  const int bh = blockIdx.x;
  const int idx = blockIdx.y * 256 + threadIdx.x;          // f*64 + d, 0..16383
  float s = 0.f;
  for (int c = 0; c < 16; ++c) s += ctx_part[((long)c * 16 + bh) * 16384 + idx];
  const int f = idx >> 6, d = idx & 63;
  ctxT[(long)bh * 16384 + d * 256 + f] = s;
  if (blockIdx.y == 0) {
    float t = 0.f;
    for (int c = 0; c < 16; ++c) t += ksum_part[((long)c * 16 + bh) * 256 + threadIdx.x];
    ksum[bh * 256 + threadIdx.x] = t;
  }
}

// ---------------------------------------------------------------------------
// K4: QpT = proj @ Q^T (scaled, exp) -> sQp[n][f] via packed writes;
//     denom = Qp . (ksum+eps); out = (Qp @ context) / denom -> out_attn bf16
// ---------------------------------------------------------------------------
__global__ __launch_bounds__(256) void k_qout(const bf16_t* __restrict__ qkv,
                                              const float* __restrict__ proj,
                                              const float* __restrict__ ctxT,
                                              const float* __restrict__ ksum,
                                              bf16_t* __restrict__ outA) {
  const int bh = blockIdx.x;
  const long n0 = (long)blockIdx.y * 128;
  const int tid = threadIdx.x, lane = tid & 31, wave = tid >> 5;
  const int wm = wave & 3, wn = wave >> 2;
  const int h = bh & 7;
  const long rowbase = (long)(bh >> 3) * 16384;
  const bf16_t* Qg = qkv + rowbase * 1536 + h * 64;

  __shared__ __attribute__((aligned(128))) bf16_t sProj[256 * 64];
  __shared__ __attribute__((aligned(128))) bf16_t sA[128 * 64];     // Q tile [n][d]
  __shared__ __attribute__((aligned(128))) bf16_t sQp[128 * 256];   // Qp [n][f]
  __shared__ __attribute__((aligned(128))) bf16_t sCT[64 * 256];    // context^T [d][f]
  __shared__ float sKs[256];
  __shared__ float sDen[128];

  // Q tile via async DMA (pure bf16 copy)
  const unsigned sA_off = lds_off_u32(sA);
  for (int idx = tid; idx < 128 * 8; idx += 256) {
    int r = idx >> 3, c8 = (idx & 7) * 8;
    async_g2l_b128(sA_off + (unsigned)(r * 64 + c8) * 2, Qg + (n0 + r) * 1536 + c8);
  }
  for (int idx = tid; idx < 256 * 16; idx += 256) {
    int r = idx >> 4, c4 = (idx & 15) * 4;
    cvt4(sProj + r * 64 + c4, *(const float4*)(proj + r * 64 + c4));
  }
  for (int idx = tid; idx < 64 * 64; idx += 256) {         // contiguous ctxT staging
    int r = idx >> 6, c4 = (idx & 63) * 4;
    cvt4(sCT + r * 256 + c4, *(const float4*)(ctxT + (long)bh * 16384 + r * 256 + c4));
  }
  sKs[tid] = ksum[bh * 256 + tid] + 1e-6f;
  wait_async0();
  __syncthreads();

  // stage 1 (swapped orientation): C[f][n] = proj @ Q^T  (M=256 f, N=128 n, K=64)
  // waves 4(f) x 2(n), wave tile 64f x 64n
  v8f e[4][4];
#pragma unroll
  for (int i = 0; i < 4; ++i)
#pragma unroll
    for (int j = 0; j < 4; ++j) vzero(e[i][j]);
#pragma unroll
  for (int k0 = 0; k0 < 64; k0 += 32) {
    v16bf a[4], b[4];
#pragma unroll
    for (int i = 0; i < 4; ++i) a[i] = frag_a(sProj + (wm * 64 + i * 16) * 64 + k0, 64, lane);
#pragma unroll
    for (int j = 0; j < 4; ++j) b[j] = frag_b(sA + (wn * 64 + j * 16) * 64 + k0, 64, lane);
#pragma unroll
    for (int i = 0; i < 4; ++i)
#pragma unroll
      for (int j = 0; j < 4; ++j) e[i][j] = WMMA_BF16(a[i], b[j], e[i][j]);
  }
  const float scale = 0.125f;  // 64^-0.5
#pragma unroll
  for (int i = 0; i < 4; ++i)
#pragma unroll
    for (int j = 0; j < 4; ++j) {
      int fb = wm * 64 + i * 16 + ((lane >> 4) << 3);      // 8 consecutive f per lane
      int n  = wn * 64 + j * 16 + (lane & 15);
      v8bf t;
#pragma unroll
      for (int r = 0; r < 8; ++r) t[r] = (bf16_t)__expf(e[i][j][r] * scale);
      *(v8bf*)(sQp + n * 256 + fb) = t;                    // packed b128 write
    }
  __syncthreads();

  if (tid < 128) {
    float s = 0.f;
    const bf16_t* p = sQp + tid * 256;
    for (int q = 0; q < 256; q += 8) {
      v8bf v = *(const v8bf*)(p + q);
#pragma unroll
      for (int u = 0; u < 8; ++u) s += (float)v[u] * sKs[q + u];
    }
    sDen[tid] = 1.f / s;
  }
  __syncthreads();

  // stage 2: out = Qp @ context (waves 4(n) x 2(d), wave tile 32n x 32d, K=256)
  v8f o[2][2];
#pragma unroll
  for (int i = 0; i < 2; ++i)
#pragma unroll
    for (int j = 0; j < 2; ++j) vzero(o[i][j]);
#pragma unroll
  for (int k0 = 0; k0 < 256; k0 += 32) {
    v16bf a[2], b[2];
#pragma unroll
    for (int i = 0; i < 2; ++i) a[i] = frag_a(sQp + (wm * 32 + i * 16) * 256 + k0, 256, lane);
#pragma unroll
    for (int j = 0; j < 2; ++j) b[j] = frag_b(sCT + (wn * 32 + j * 16) * 256 + k0, 256, lane);
#pragma unroll
    for (int i = 0; i < 2; ++i)
#pragma unroll
      for (int j = 0; j < 2; ++j) o[i][j] = WMMA_BF16(a[i], b[j], o[i][j]);
  }
#pragma unroll
  for (int i = 0; i < 2; ++i)
#pragma unroll
    for (int j = 0; j < 2; ++j)
#pragma unroll
      for (int r = 0; r < 8; ++r) {
        int n = wm * 32 + i * 16 + r + ((lane >> 4) << 3);
        int d = wn * 32 + j * 16 + (lane & 15);
        outA[(rowbase + n0 + n) * 512 + h * 64 + d] = (bf16_t)(o[i][j][r] * sDen[n]);
      }
}

// ---------------------------------------------------------------------------
// K5: out[32768 x 512] = out_attn(bf16) @ W_out^T + b_out  (fp32 output)
// ---------------------------------------------------------------------------
__global__ __launch_bounds__(256) void k_final(const bf16_t* __restrict__ A,
                                               const float* __restrict__ W,
                                               const float* __restrict__ bias,
                                               float* __restrict__ Y) {
  const int tid = threadIdx.x, lane = tid & 31, wave = tid >> 5;
  const int wm = wave & 3, wn = wave >> 2;
  const long m0 = (long)blockIdx.y * 128;
  const int  n0 = blockIdx.x * 128;
  __shared__ __attribute__((aligned(128))) bf16_t sA[128 * 32];
  __shared__ __attribute__((aligned(128))) bf16_t sB[128 * 32];
  const unsigned sA_off = lds_off_u32(sA);
  v8f acc[2][4];
#pragma unroll
  for (int i = 0; i < 2; ++i)
#pragma unroll
    for (int j = 0; j < 4; ++j) vzero(acc[i][j]);

  for (int k0 = 0; k0 < 512; k0 += 32) {
    __syncthreads();
    for (int idx = tid; idx < 128 * 4; idx += 256) {       // bf16 A tile via async DMA
      int r = idx >> 2, c8 = (idx & 3) * 8;
      async_g2l_b128(sA_off + (unsigned)(r * 32 + c8) * 2, A + (m0 + r) * 512 + k0 + c8);
    }
    for (int idx = tid; idx < 128 * 8; idx += 256) {
      int r = idx >> 3, c4 = (idx & 7) * 4;
      cvt4(sB + r * 32 + c4, *(const float4*)(W + (long)(n0 + r) * 512 + k0 + c4));
    }
    if (k0 + 32 < 512) {
      if (tid < 128) __builtin_prefetch(A + (m0 + tid) * 512 + k0 + 32, 0, 1);
      else           __builtin_prefetch(W + (long)(n0 + tid - 128) * 512 + k0 + 32, 0, 1);
    }
    wait_async0();
    __syncthreads();
    v16bf a[2], b[4];
#pragma unroll
    for (int i = 0; i < 2; ++i) a[i] = frag_a(sA + (wm * 32 + i * 16) * 32, 32, lane);
#pragma unroll
    for (int j = 0; j < 4; ++j) b[j] = frag_b(sB + (wn * 64 + j * 16) * 32, 32, lane);
#pragma unroll
    for (int i = 0; i < 2; ++i)
#pragma unroll
      for (int j = 0; j < 4; ++j) acc[i][j] = WMMA_BF16(a[i], b[j], acc[i][j]);
  }
#pragma unroll
  for (int i = 0; i < 2; ++i)
#pragma unroll
    for (int j = 0; j < 4; ++j)
#pragma unroll
      for (int r = 0; r < 8; ++r) {
        long row = m0 + wm * 32 + i * 16 + r + ((lane >> 4) << 3);
        int  col = n0 + wn * 64 + j * 16 + (lane & 15);
        Y[row * 512 + col] = acc[i][j][r] + bias[col];
      }
}

// ---------------------------------------------------------------------------
extern "C" void kernel_launch(void* const* d_in, const int* in_sizes, int n_in,
                              void* d_out, int out_size, void* d_ws, size_t ws_size,
                              hipStream_t stream) {
  (void)in_sizes; (void)n_in; (void)out_size; (void)ws_size;
  const float* x    = (const float*)d_in[0];
  const float* Wqkv = (const float*)d_in[1];
  const float* proj = (const float*)d_in[2];
  const float* Wout = (const float*)d_in[3];
  const float* bout = (const float*)d_in[4];
  float* out = (float*)d_out;

  char* ws = (char*)d_ws;
  size_t off = 0;
  bf16_t* qkv = (bf16_t*)(ws + off);       off += (size_t)32768 * 1536 * 2;    // 96 MB (q,k used)
  bf16_t* vT  = (bf16_t*)(ws + off);       off += (size_t)16 * 64 * 16384 * 2; // 32 MB
  float* ctx_part  = (float*)(ws + off);   off += (size_t)16 * 16 * 16384 * 4; // 16 MB
  float* ksum_part = (float*)(ws + off);   off += (size_t)16 * 16 * 256 * 4;
  float* ctxT      = (float*)(ws + off);   off += (size_t)16 * 16384 * 4;
  float* ksum      = (float*)(ws + off);   off += (size_t)16 * 256 * 4;
  bf16_t* outA     = (bf16_t*)(ws + off);  off += (size_t)32768 * 512 * 2;     // 32 MB

  dim3 blk(256);
  k_qkv  <<<dim3(12, 256), blk, 0, stream>>>(x, Wqkv, qkv, vT);
  k_kfeat<<<dim3(16, 16),  blk, 0, stream>>>(qkv, vT, proj, ctx_part, ksum_part);
  k_reduce<<<dim3(16, 64), blk, 0, stream>>>(ctx_part, ksum_part, ctxT, ksum);
  k_qout <<<dim3(16, 128), blk, 0, stream>>>(qkv, proj, ctxT, ksum, outA);
  k_final<<<dim3(4, 256),  blk, 0, stream>>>(outA, Wout, bout, out);
}
